// FastAttention_23450521436759
// MI455X (gfx1250) — matrix operand order repaired and checked by reading, not compiled, not run on hardware
//
#include <hip/hip_runtime.h>
#include <hip/hip_bf16.h>

// ---------------------------------------------------------------------------
// Performer / FAVOR+ attention block for MI455X (gfx1250, wave32, WMMA).
// All heavy contractions run through a generic 128x128x32 WMMA-f16 GEMM
// (f32 accumulate). Elementwise feature-map / reductions are VALU kernels.
// ---------------------------------------------------------------------------

typedef __attribute__((ext_vector_type(16))) _Float16 v16h;
typedef __attribute__((ext_vector_type(8)))  _Float16 half8;
typedef __attribute__((ext_vector_type(8)))  float    v8f;

constexpr int BMT = 128;   // block tile M
constexpr int BNT = 128;   // block tile N
constexpr int BKT = 32;    // block tile K
constexpr int LDT = 40;    // LDS row stride in halves (32 + 8 pad, keeps 16B align)

// ---- WMMA fragment loaders (per CDNA5 ISA 7.12.2 16-bit layouts) ----------
// A 16x32 (MxK): lane 0-15 -> M=lane, K={0..7}+8h and {16..23}+8h (h=lane>>4)
__device__ inline v16h frag_a(const _Float16* __restrict__ p) {
  int lane = threadIdx.x & 31;
  int row  = lane & 15;
  int half = lane >> 4;
  const _Float16* q = p + row * LDT + half * 8;
  union { v16h v; half8 h[2]; } u;
  u.h[0] = *(const half8*)(q);
  u.h[1] = *(const half8*)(q + 16);
  return u.v;
}
// B 32x16 (KxN) staged col-major in LDS ([n][k]): lane 0-15 -> N=lane,
// K={0..7}+8h and {16..23}+8h
__device__ inline v16h frag_b(const _Float16* __restrict__ p) {
  int lane = threadIdx.x & 31;
  int col  = lane & 15;
  int half = lane >> 4;
  const _Float16* q = p + col * LDT + half * 8;
  union { v16h v; half8 h[2]; } u;
  u.h[0] = *(const half8*)(q);
  u.h[1] = *(const half8*)(q + 16);
  return u.v;
}

// ---------------------------------------------------------------------------
// Generic batched WMMA GEMM:  C[z] = A[z] (MxK) * B[z] (KxN)
//   z = blockIdx.z ; b = z / H ; h = z % H ; base = s_b*b + s_h*h
//   TA: A stored transposed (addr = k*lda + row), used for K'^T V
//   OH: write f16 output, else f32
//   TC: transposed store (addr = col*ldc + row), used for final (b,c,l)
//   RS: per-row scale (d_inv) applied in epilogue
// All of M,N multiples of 128 and K multiple of 32 by construction.
// ---------------------------------------------------------------------------
template<bool TA, bool OH, bool TC, bool RS>
__global__ __launch_bounds__(256) void wmma_gemm(
    const _Float16* __restrict__ A, long a_sb, long a_sh, int lda,
    const _Float16* __restrict__ B, long b_sb, long b_sh, int ldb,
    void* __restrict__ Cv, long c_sb, long c_sh, int ldc,
    const float* __restrict__ rowscale,
    int M, int N, int K, int H)
{
  __shared__ __align__(16) _Float16 As[BMT * LDT];
  __shared__ __align__(16) _Float16 Bs[BNT * LDT];

  int z  = blockIdx.z;
  int bb = z / H;
  int hh = z % H;
  const _Float16* Ab = A + a_sb * bb + a_sh * hh;
  const _Float16* Bb = B + b_sb * bb + b_sh * hh;

  int n0 = blockIdx.x * BNT;
  int m0 = blockIdx.y * BMT;
  int tid = threadIdx.x;
  int wid = tid >> 5;
  int wm  = wid & 3;   // 4 waves along M (32 rows each)
  int wn  = wid >> 2;  // 2 waves along N (64 cols each)

  v8f zero = {0.f,0.f,0.f,0.f,0.f,0.f,0.f,0.f};
  v8f acc[2][4];
#pragma unroll
  for (int mi = 0; mi < 2; ++mi)
#pragma unroll
    for (int ni = 0; ni < 4; ++ni) acc[mi][ni] = zero;

  for (int k0 = 0; k0 < K; k0 += BKT) {
    // ---- stage A tile (128 x 32) ----
    if (!TA) {
#pragma unroll
      for (int i = 0; i < 2; ++i) {
        int lin = tid + i * 256;            // 0..511
        int r   = lin >> 2;                 // 0..127
        int c8  = (lin & 3) << 3;           // 0,8,16,24
        half8 val = *(const half8*)(Ab + (long)(m0 + r) * lda + (k0 + c8));
        *(half8*)(As + r * LDT + c8) = val;
      }
    } else {
#pragma unroll
      for (int i = 0; i < 2; ++i) {
        int lin = tid + i * 256;
        int k   = lin >> 4;                 // 0..31
        int r8  = (lin & 15) << 3;          // 0..120
        half8 val = *(const half8*)(Ab + (long)(k0 + k) * lda + (m0 + r8));
#pragma unroll
        for (int j = 0; j < 8; ++j) As[(r8 + j) * LDT + k] = val[j];
      }
    }
    // ---- stage B tile (32 x 128) into col-major LDS [n][k] ----
#pragma unroll
    for (int i = 0; i < 2; ++i) {
      int lin = tid + i * 256;
      int k   = lin >> 4;                   // 0..31
      int n8  = (lin & 15) << 3;            // 0..120
      half8 val = *(const half8*)(Bb + (long)(k0 + k) * ldb + (n0 + n8));
#pragma unroll
      for (int j = 0; j < 8; ++j) Bs[(n8 + j) * LDT + k] = val[j];
    }
    __syncthreads();

    v16h af[2], bf[4];
#pragma unroll
    for (int mi = 0; mi < 2; ++mi)
      af[mi] = frag_a(As + (wm * 32 + mi * 16) * LDT);
#pragma unroll
    for (int ni = 0; ni < 4; ++ni)
      bf[ni] = frag_b(Bs + (wn * 64 + ni * 16) * LDT);
#pragma unroll
    for (int mi = 0; mi < 2; ++mi)
#pragma unroll
      for (int ni = 0; ni < 4; ++ni)
        acc[mi][ni] = __builtin_amdgcn_wmma_f32_16x16x32_f16(
            false, af[mi], false, bf[ni], (short)0, acc[mi][ni], false, false);
    __syncthreads();
  }

  // ---- epilogue: D layout = (M = v + 8*half, N = lane&15) ----
  long cbase = c_sb * bb + c_sh * hh;
  int lane = tid & 31;
  int coll = lane & 15;
  int half = lane >> 4;
  const float* rsp = RS ? (rowscale + (long)z * M) : nullptr;
#pragma unroll
  for (int mi = 0; mi < 2; ++mi) {
#pragma unroll
    for (int ni = 0; ni < 4; ++ni) {
      int rbase = m0 + wm * 32 + mi * 16 + half * 8;
      int c     = n0 + wn * 64 + ni * 16 + coll;
#pragma unroll
      for (int v = 0; v < 8; ++v) {
        int r = rbase + v;
        float val = acc[mi][ni][v];
        if (RS) val *= rsp[r];
        long addr = TC ? (cbase + (long)c * ldc + r)
                       : (cbase + (long)r * ldc + c);
        if (OH) ((_Float16*)Cv)[addr] = (_Float16)val;
        else    ((float*)Cv)[addr]    = val;
      }
    }
  }
}

// ---------------------------------------------------------------------------
// x (b,c,l) f32  ->  xh (b,l,c) f16   (tiled transpose)
// ---------------------------------------------------------------------------
__global__ void transpose_cast_kernel(const float* __restrict__ x,
                                      _Float16* __restrict__ xh, int C, int L) {
  __shared__ float tile[32][33];
  int b  = blockIdx.z;
  int c0 = blockIdx.y * 32;
  int l0 = blockIdx.x * 32;
  for (int j = threadIdx.y; j < 32; j += 8)
    tile[j][threadIdx.x] = x[(size_t)(b * C + c0 + j) * L + l0 + threadIdx.x];
  __syncthreads();
  for (int j = threadIdx.y; j < 32; j += 8)
    xh[(size_t)(b * L + l0 + j) * C + c0 + threadIdx.x] =
        (_Float16)tile[threadIdx.x][j];
}

// out[(h*C + c)*Mf + m] = dn * sum_e proj[m,e] * W[h,e,c]   (fold proj into W)
__global__ void combine_kernel(const float* __restrict__ proj,
                               const float* __restrict__ W,
                               _Float16* __restrict__ out,
                               float dn, int Mf, int E, int C) {
  int m = threadIdx.x;
  int c = blockIdx.x;
  int h = blockIdx.y;
  float s = 0.f;
  for (int e = 0; e < E; ++e)
    s += proj[m * E + e] * W[((size_t)h * E + e) * C + c];
  out[((size_t)h * C + c) * Mf + m] = (_Float16)(dn * s);
}

// tvT[h,ci,co] = tovalue[h,co,ci]  (f16)
__global__ void transpose_w_kernel(const float* __restrict__ W,
                                   _Float16* __restrict__ out, int C) {
  int co = threadIdx.x;
  int ci = blockIdx.x;
  int h  = blockIdx.y;
  out[((size_t)h * C + ci) * C + co] =
      (_Float16)W[((size_t)h * C + co) * C + ci];
}

// W2[(h*E2+e)*C + c] = toout[h,c,e]  (f16)
__global__ void build_w2_kernel(const float* __restrict__ toout,
                                _Float16* __restrict__ w2, int C, int E2) {
  int c = threadIdx.x;
  int e = blockIdx.x;
  int h = blockIdx.y;
  w2[((size_t)h * E2 + e) * C + c] =
      (_Float16)toout[((size_t)h * C + c) * E2 + e];
}

// diag[b,h,l] = coef * || W[h] @ x[b,:,l] ||^2     (E = 64 fixed)
__global__ __launch_bounds__(256) void diag_kernel(
    const float* __restrict__ x, const float* __restrict__ W,
    float* __restrict__ diag, float coef, int L, int C) {
  int l = blockIdx.x * blockDim.x + threadIdx.x;
  int h = blockIdx.y, b = blockIdx.z;
  int H = gridDim.y;
  const float* Wh = W + (size_t)h * 64 * C;
  const float* xb = x + (size_t)b * C * L;
  float acc[64];
#pragma unroll
  for (int e = 0; e < 64; ++e) acc[e] = 0.f;
  for (int c = 0; c < C; ++c) {
    float xv = xb[(size_t)c * L + l];
#pragma unroll
    for (int e = 0; e < 64; ++e) acc[e] = fmaf(Wh[e * C + c], xv, acc[e]);
  }
  float s = 0.f;
#pragma unroll
  for (int e = 0; e < 64; ++e) s += acc[e] * acc[e];
  diag[((size_t)(b * H + h)) * L + l] = coef * s;
}

// q'[row,:] = ratio * (exp(dd - diag[row] - rowmax) + eps)   (one wave / row)
__global__ __launch_bounds__(256) void qprime_kernel(
    const float* __restrict__ dd, const float* __restrict__ diag,
    _Float16* __restrict__ qp, float ratio, float eps, int Mf) {
  long row = (long)blockIdx.x * 8 + (threadIdx.x >> 5);
  int lane = threadIdx.x & 31;
  const float* dr = dd + row * Mf;
  float mx = -3.4e38f;
  for (int i = lane; i < Mf; i += 32) mx = fmaxf(mx, dr[i]);
#pragma unroll
  for (int off = 16; off > 0; off >>= 1)
    mx = fmaxf(mx, __shfl_xor(mx, off, 32));
  float dg = diag[row];
  for (int i = lane; i < Mf; i += 32)
    qp[row * Mf + i] = (_Float16)(ratio * (__expf(dr[i] - dg - mx) + eps));
}

// stab[bh] = max over (l,m) of dd_k[bh]
__global__ __launch_bounds__(256) void kmax_kernel(
    const float* __restrict__ dd, float* __restrict__ stab, long n) {
  long base = (long)blockIdx.x * n;
  float mx = -3.4e38f;
  for (long i = threadIdx.x; i < n; i += 256) mx = fmaxf(mx, dd[base + i]);
  __shared__ float red[256];
  red[threadIdx.x] = mx;
  __syncthreads();
  for (int s = 128; s > 0; s >>= 1) {
    if (threadIdx.x < s)
      red[threadIdx.x] = fmaxf(red[threadIdx.x], red[threadIdx.x + s]);
    __syncthreads();
  }
  if (threadIdx.x == 0) stab[blockIdx.x] = red[0];
}

// k'[idx] = ratio * (exp(dd - diag[row] - stab[bh]) + eps)
__global__ __launch_bounds__(256) void kprime_kernel(
    const float* __restrict__ dd, const float* __restrict__ diag,
    const float* __restrict__ stab, _Float16* __restrict__ kp,
    float ratio, float eps, int Mf, int L) {
  long idx = (long)blockIdx.x * 256 + threadIdx.x;
  long row = idx / Mf;
  long bh  = row / L;
  kp[idx] = (_Float16)(ratio * (__expf(dd[idx] - diag[row] - stab[bh]) + eps));
}

// kcum[bh,m] = sum_l k'[bh,l,m]
__global__ __launch_bounds__(256) void kcum_kernel(
    const _Float16* __restrict__ kp, float* __restrict__ kcum, int Mf, int L) {
  int m  = threadIdx.x;
  int bh = blockIdx.x;
  const _Float16* base = kp + (size_t)bh * L * Mf;
  float s = 0.f;
  for (int l = 0; l < L; ++l) s += (float)base[(size_t)l * Mf + m];
  kcum[(size_t)bh * Mf + m] = s;
}

// dinv[row] = 1 / dot(q'[row,:], kcum[bh,:])   (one wave / row)
__global__ __launch_bounds__(256) void dinv_kernel(
    const _Float16* __restrict__ qp, const float* __restrict__ kcum,
    float* __restrict__ dinv, int Mf, int L) {
  long row = (long)blockIdx.x * 8 + (threadIdx.x >> 5);
  int lane = threadIdx.x & 31;
  long bh  = row / L;
  const _Float16* qr = qp + row * Mf;
  const float* kc = kcum + bh * Mf;
  float s = 0.f;
  for (int i = lane; i < Mf; i += 32) s += (float)qr[i] * kc[i];
#pragma unroll
  for (int off = 16; off > 0; off >>= 1) s += __shfl_xor(s, off, 32);
  if (lane == 0) dinv[row] = 1.0f / s;
}

// ---------------------------------------------------------------------------
extern "C" void kernel_launch(void* const* d_in, const int* in_sizes, int n_in,
                              void* d_out, int out_size, void* d_ws, size_t ws_size,
                              hipStream_t stream) {
  const float* x        = (const float*)d_in[0];
  const float* tokey    = (const float*)d_in[1];
  const float* toquery  = (const float*)d_in[2];
  const float* tovalue  = (const float*)d_in[3];
  const float* toout    = (const float*)d_in[4];
  const float* proj     = (const float*)d_in[5];
  float* out = (float*)d_out;

  constexpr int B = 8, C = 256, L = 4096, H = 8, E = 64, Mf = 256;
  constexpr float dn    = 0.35355339059327379f;  // 64^-0.25
  constexpr float coef  = 0.0625f;               // 0.5 * dn^2
  constexpr float ratio = 0.0625f;               // 256^-0.5
  constexpr float eps   = 1e-4f;

  // ---- workspace carve-up (~0.96 GB, buffers reused where lifetimes allow)
  char* ws = (char*)d_ws;
  size_t off = 0;
  auto alloc = [&](size_t bytes) -> void* {
    void* p = ws + off;
    off = (off + bytes + 255) & ~(size_t)255;
    return p;
  };
  _Float16* xh    = (_Float16*)alloc((size_t)B * L * C * 2);      // 16.8 MB
  _Float16* aqT   = (_Float16*)alloc((size_t)H * C * Mf * 2);     // 1 MB
  _Float16* akT   = (_Float16*)alloc((size_t)H * C * Mf * 2);
  _Float16* tvT   = (_Float16*)alloc((size_t)H * C * C * 2);
  _Float16* w2    = (_Float16*)alloc((size_t)H * 256 * C * 2);
  float*    ddq   = (float*)alloc((size_t)B * H * L * Mf * 4);    // 268 MB
  float*    ddk   = (float*)alloc((size_t)B * H * L * Mf * 4);    // 268 MB
  _Float16* qp    = (_Float16*)alloc((size_t)B * H * L * Mf * 2); // 134 MB
  _Float16* kp    = (_Float16*)alloc((size_t)B * H * L * Mf * 2); // 134 MB
  _Float16* vh    = (_Float16*)alloc((size_t)B * H * L * C * 2);  // 134 MB
  float*    diagq = (float*)alloc((size_t)B * H * L * 4);
  float*    diagk = (float*)alloc((size_t)B * H * L * 4);
  float*    dinv  = (float*)alloc((size_t)B * H * L * 4);
  float*    kcum  = (float*)alloc((size_t)B * H * Mf * 4);
  float*    stab  = (float*)alloc((size_t)B * H * 4);
  _Float16* out2  = (_Float16*)ddq;  // ddq dead after qprime -> reuse for out
  _Float16* ctx   = (_Float16*)ddk;  // ddk dead after kprime -> reuse for ctx

  // 1) preprocessing
  transpose_cast_kernel<<<dim3(L / 32, C / 32, B), dim3(32, 8), 0, stream>>>(x, xh, C, L);
  combine_kernel<<<dim3(C, H), Mf, 0, stream>>>(proj, toquery, aqT, dn, Mf, E, C);
  combine_kernel<<<dim3(C, H), Mf, 0, stream>>>(proj, tokey,   akT, dn, Mf, E, C);
  transpose_w_kernel<<<dim3(C, H), C, 0, stream>>>(tovalue, tvT, C);
  build_w2_kernel<<<dim3(256, H), C, 0, stream>>>(toout, w2, C, 256);
  diag_kernel<<<dim3(L / 256, H, B), 256, 0, stream>>>(x, toquery, diagq, coef, L, C);
  diag_kernel<<<dim3(L / 256, H, B), 256, 0, stream>>>(x, tokey,   diagk, coef, L, C);

  // 2) dd_q / dd_k : (l x m) = xh (l x c) @ A^T (c x m), per (b,h)
  wmma_gemm<false, false, false, false><<<dim3(Mf / BNT, L / BMT, B * H), 256, 0, stream>>>(
      xh,  (long)L * C, 0,            C,
      aqT, 0,           (long)C * Mf, Mf,
      ddq, (long)H * L * Mf, (long)L * Mf, Mf,
      nullptr, L, Mf, C, H);
  wmma_gemm<false, false, false, false><<<dim3(Mf / BNT, L / BMT, B * H), 256, 0, stream>>>(
      xh,  (long)L * C, 0,            C,
      akT, 0,           (long)C * Mf, Mf,
      ddk, (long)H * L * Mf, (long)L * Mf, Mf,
      nullptr, L, Mf, C, H);

  // 3) feature maps
  qprime_kernel<<<(B * H * L) / 8, 256, 0, stream>>>(ddq, diagq, qp, ratio, eps, Mf);
  kmax_kernel<<<B * H, 256, 0, stream>>>(ddk, stab, (long)L * Mf);
  kprime_kernel<<<(size_t)B * H * L * Mf / 256, 256, 0, stream>>>(ddk, diagk, stab, kp, ratio, eps, Mf, L);
  kcum_kernel<<<B * H, 256, 0, stream>>>(kp, kcum, Mf, L);

  // 4) value projection: vh (l x c) = xh @ tvT, per (b,h), f16 out
  wmma_gemm<false, true, false, false><<<dim3(C / BNT, L / BMT, B * H), 256, 0, stream>>>(
      xh,  (long)L * C, 0,           C,
      tvT, 0,           (long)C * C, C,
      vh,  (long)H * L * C, (long)L * C, C,
      nullptr, L, C, C, H);

  // 5) context (m x c) = k'^T (m x l) @ vh (l x c), per (b,h), f16 out
  wmma_gemm<true, true, false, false><<<dim3(C / BNT, Mf / BMT, B * H), 256, 0, stream>>>(
      kp, (long)H * L * Mf, (long)L * Mf, Mf,
      vh, (long)H * L * C,  (long)L * C,  C,
      ctx, (long)H * Mf * C, (long)Mf * C, C,
      nullptr, Mf, C, L, H);

  // 6) d_inv
  dinv_kernel<<<(B * H * L) / 8, 256, 0, stream>>>(qp, kcum, dinv, Mf, L);

  // 7) out2 (b,l,h*256+e) = dinv .* (q' @ ctx), per (b,h), f16 out
  wmma_gemm<false, true, false, true><<<dim3(C / BNT, L / BMT, B * H), 256, 0, stream>>>(
      qp,  (long)H * L * Mf, (long)L * Mf, Mf,
      ctx, (long)H * Mf * C, (long)Mf * C, C,
      out2, (long)L * H * C, 256, H * C,
      dinv, L, C, Mf, H);

  // 8) final (b,c,l) = out2 (l x 2048) @ W2 (2048 x 256), transposed f32 store
  wmma_gemm<false, false, true, false><<<dim3(C / BNT, L / BMT, B), 256, 0, stream>>>(
      out2, (long)L * H * C, 0, H * C,
      w2,   0, 0, C,
      out,  (long)C * L, 0, L,
      nullptr, L, C, H * C, 1);
}
